// YoloXAssocHead_56014963475156
// MI455X (gfx1250) — matrix operand order repaired and checked by reading, not compiled, not run on hardware
//
#include <hip/hip_runtime.h>
#include <hip/hip_bf16.h>
#include <math.h>

typedef __attribute__((ext_vector_type(16))) _Float16 v16h;
typedef __attribute__((ext_vector_type(8)))  float    v8f;

#define NANCHORS 21504
#define NPAIR    90000
#define MTILE    128     // 8 waves * 16 rows
#define MAINROWS 89984   // 703 * 128; tail block covers rows 89984..89999

__device__ __forceinline__ float sigmoidf(float x) { return 1.0f / (1.0f + expf(-x)); }

union H16 { v16h v; unsigned int u[8]; };

// ---------------------------------------------------------------------------
// zero the BN-stats accumulators: 16 images * 4 stages * 128 floats
// ---------------------------------------------------------------------------
__global__ void zero_stats_kernel(float* stats) {
    for (int i = threadIdx.x; i < 16 * 4 * 128; i += blockDim.x) stats[i] = 0.0f;
}

// ---------------------------------------------------------------------------
// Decode one FPN level: sigmoid(cls)*sigmoid(obj) with NCHW->NHWC transpose,
// box decode (xy*s + centers, exp(wh)*s). 64 positions per block.
// ---------------------------------------------------------------------------
__global__ __launch_bounds__(256)
void decode_kernel(const float* __restrict__ cls, const float* __restrict__ loc,
                   const float* __restrict__ obj, float* __restrict__ scores,
                   float* __restrict__ boxes, int H, float s, int aoff) {
    __shared__ float sObj[64];
    __shared__ float sS[80 * 64];
    const int tid = threadIdx.x;
    const int b = blockIdx.y;
    const int posBase = blockIdx.x * 64;
    const int HW = H * H;

    if (tid < 64) sObj[tid] = sigmoidf(obj[(long)b * HW + posBase + tid]);
    __syncthreads();

    const int p = tid & 63;
    for (int c = tid >> 6; c < 80; c += 4) {
        float x = cls[((long)b * 80 + c) * HW + posBase + p];
        sS[c * 64 + p] = sigmoidf(x) * sObj[p];
    }
    __syncthreads();

    const long obase = ((long)b * NANCHORS + aoff + posBase) * 80;
    for (int i = tid; i < 80 * 64; i += 256) {
        int pp = i / 80, c = i % 80;
        scores[obase + (long)pp * 80 + c] = sS[c * 64 + pp];
    }

    if (tid < 64) {
        int pos = posBase + tid;
        int iy = pos / H, ix = pos % H;
        float l0 = loc[((long)b * 4 + 0) * HW + pos];
        float l1 = loc[((long)b * 4 + 1) * HW + pos];
        float l2 = loc[((long)b * 4 + 2) * HW + pos];
        float l3 = loc[((long)b * 4 + 3) * HW + pos];
        long bb = ((long)b * NANCHORS + aoff + pos) * 4;
        boxes[bb + 0] = l0 * s + ((float)ix + 0.5f) * s;
        boxes[bb + 1] = l1 * s + ((float)iy + 0.5f) * s;
        boxes[bb + 2] = expf(l2) * s;
        boxes[bb + 3] = expf(l3) * s;
    }
}

// ---------------------------------------------------------------------------
// L2-normalize embeddings: one wave (32 lanes, 2 floats/lane) per row of 64.
// rows 0..4799 = det (id_embeds), 4800..9599 = ref.
// ---------------------------------------------------------------------------
__global__ __launch_bounds__(256)
void norm_embed_kernel(const float* __restrict__ de, const float* __restrict__ re,
                       float* __restrict__ dn, float* __restrict__ rn) {
    int gid = blockIdx.x * 256 + threadIdx.x;
    int wv = gid >> 5, lane = gid & 31;
    if (wv >= 9600) return;
    const float* src; float* dst; int row;
    if (wv < 4800) { src = de; dst = dn; row = wv; }
    else           { src = re; dst = rn; row = wv - 4800; }
    float2 v = ((const float2*)(src + (long)row * 64))[lane];
    float ss = v.x * v.x + v.y * v.y;
    #pragma unroll
    for (int m = 16; m >= 1; m >>= 1) ss += __shfl_xor(ss, m, 32);
    float inv = 1.0f / fmaxf(sqrtf(ss), 1e-8f);
    float2 o; o.x = v.x * inv; o.y = v.y * inv;
    ((float2*)(dst + (long)row * 64))[lane] = o;
}

// ---------------------------------------------------------------------------
// Per-box meta: center x/y and sqrt(area). rows 0..4799 det, 4800..9599 ref.
// ---------------------------------------------------------------------------
__global__ void boxmeta_kernel(const float* __restrict__ db, const float* __restrict__ rb,
                               float* __restrict__ meta) {
    int t = blockIdx.x * 256 + threadIdx.x;
    if (t >= 9600) return;
    const float* bx = (t < 4800) ? (db + (long)t * 4) : (rb + (long)(t - 4800) * 4);
    float x1 = bx[0], y1 = bx[1], x2 = bx[2], y2 = bx[3];
    meta[(long)t * 3 + 0] = (x1 + x2) * 0.5f;
    meta[(long)t * 3 + 1] = (y1 + y2) * 0.5f;
    meta[(long)t * 3 + 2] = sqrtf((x2 - x1) * (y2 - y1));
}

// ---------------------------------------------------------------------------
// Pair features for image b: cos = dn[i].rn[j], nloc = |dc-rc| / (min(shp)+eps).
// Also accumulates BN0 batch stats (sum, sumsq per channel).
// ---------------------------------------------------------------------------
__global__ __launch_bounds__(256)
void feats_kernel(const float* __restrict__ dn, const float* __restrict__ rn,
                  const float* __restrict__ meta, float* __restrict__ feats,
                  float* __restrict__ stats0, int b) {
    __shared__ float red[4];
    int tid = threadIdx.x;
    if (tid < 4) red[tid] = 0.0f;
    __syncthreads();

    int p = blockIdx.x * 256 + tid;
    float cosv = 0.0f, nloc = 0.0f;
    bool valid = p < NPAIR;
    if (valid) {
        int i = p / 300, j = p % 300;
        const float* a = dn + ((long)b * 300 + i) * 64;
        const float* c = rn + ((long)b * 300 + j) * 64;
        float s = 0.0f;
        #pragma unroll 4
        for (int k = 0; k < 64; ++k) s += a[k] * c[k];
        cosv = s;
        const float* dm = meta + (long)(b * 300 + i) * 3;
        const float* rm = meta + (long)(4800 + b * 300 + j) * 3;
        float dx = dm[0] - rm[0], dy = dm[1] - rm[1];
        float loc = sqrtf(dx * dx + dy * dy);
        nloc = loc / (fminf(dm[2], rm[2]) + 1e-6f);
        feats[(long)p * 2 + 0] = cosv;
        feats[(long)p * 2 + 1] = nloc;
    }
    float s0 = valid ? cosv : 0.0f;
    float s1 = valid ? cosv * cosv : 0.0f;
    float s2 = valid ? nloc : 0.0f;
    float s3 = valid ? nloc * nloc : 0.0f;
    #pragma unroll
    for (int m = 16; m >= 1; m >>= 1) {
        s0 += __shfl_xor(s0, m, 32); s1 += __shfl_xor(s1, m, 32);
        s2 += __shfl_xor(s2, m, 32); s3 += __shfl_xor(s3, m, 32);
    }
    if ((tid & 31) == 0) {
        atomicAdd(&red[0], s0); atomicAdd(&red[1], s1);
        atomicAdd(&red[2], s2); atomicAdd(&red[3], s3);
    }
    __syncthreads();
    // stats layout: [ch 0..63] = sum, [64 + ch] = sumsq
    if (tid == 0) { atomicAdd(&stats0[0], red[0]);  atomicAdd(&stats0[64 + 0], red[1]); }
    if (tid == 1) { atomicAdd(&stats0[1], red[2]);  atomicAdd(&stats0[64 + 1], red[3]); }
}

// ---------------------------------------------------------------------------
// Fused BN(train-mode batch stats) + Linear + (ReLU) GEMM, WMMA f16 -> f32.
// All shapes/flags are compile-time. TAIL=false instances (703 of 704 blocks)
// have NO bounds checks: straight-line ds_load_b32 fragment builds, v_wmma,
// and unconditional stores through one per-lane base pointer with immediate
// offsets (r*NOUT elements, compile-time). The single TAIL=true block handles
// rows 89984..89999 with per-element guards.
//   A: per-wave 16 rows staged in LDS row-major [r][k] (BN applied on load).
//   B: weights staged TRANSPOSED [n][k] so fragment pairs are contiguous b32.
// ---------------------------------------------------------------------------
template <int KIN, int NOUT, bool RELU, bool INF32, bool F16OUT, bool STATS, bool TAIL>
__global__ __launch_bounds__(256)
void aff_gemm_kernel(const float* __restrict__ xf, const _Float16* __restrict__ xh,
                     const float* __restrict__ W, const float* __restrict__ bias,
                     const float* __restrict__ g, const float* __restrict__ be,
                     const float* __restrict__ statsIn, float* statsOut,
                     _Float16* __restrict__ outh, float* __restrict__ outf,
                     int M, int rowOff) {
    __shared__ _Float16 sWt[64 * 64];       // [n][k] transposed, stride 64
    __shared__ _Float16 sA[8][16 * 64];     // per-wave [r][k], stride 64
    __shared__ float sScale[64], sShift[64], sBias[64];
    __shared__ float sSum[64], sSumSq[64];

    const int tid = threadIdx.x;
    const int wave = tid >> 5;
    const int lane = tid & 31;
    const int halfsel = lane >> 4;
    const int l16 = lane & 15;
    const int rowBase = rowOff + blockIdx.x * MTILE + wave * 16;

    // phase 1: BN scale/shift from batch stats, bias, zero stat reducers
    if (tid < KIN) {
        float s = statsIn[tid], q = statsIn[64 + tid];
        float mean = s / (float)M;
        float var = q / (float)M - mean * mean;
        float sc = g[tid] * rsqrtf(var + 1e-5f);
        sScale[tid] = sc;
        sShift[tid] = be[tid] - mean * sc;
    }
    if (tid < NOUT) { sBias[tid] = bias[tid]; sSum[tid] = 0.0f; sSumSq[tid] = 0.0f; }
    __syncthreads();

    // phase 2: stage weights transposed (f32 -> f16), stage BN(x) rows
    #pragma unroll
    for (int i = tid; i < KIN * NOUT; i += 256) {
        int k = i / NOUT, n = i % NOUT;
        sWt[n * 64 + k] = (_Float16)W[i];
    }
    #pragma unroll
    for (int idx = lane; idx < 16 * KIN; idx += 32) {
        int r = idx / KIN, c = idx % KIN;
        int row = rowBase + r;
        float v = 0.0f;
        if (!TAIL || row < M)
            v = INF32 ? xf[(long)row * KIN + c] : (float)xh[(long)row * KIN + c];
        v = v * sScale[c] + sShift[c];
        sA[wave][r * 64 + c] = (_Float16)v;
    }
    __syncthreads();

    // A fragments (ISA 16-bit A 16x32 layout); KIN is even so pairs are whole
    H16 a0, a1;
    #pragma unroll
    for (int v = 0; v < 8; ++v) {
        const int kl = (v < 4 ? 2 * v : 2 * v + 8) + (halfsel ? 8 : 0);
        a0.u[v] = (kl < KIN)      ? *(const unsigned int*)&sA[wave][l16 * 64 + kl]      : 0u;
        a1.u[v] = (kl + 32 < KIN) ? *(const unsigned int*)&sA[wave][l16 * 64 + kl + 32] : 0u;
    }

    // per-lane first output row (lane's 8 C values are rows row0 + r)
    const int row0 = rowBase + (halfsel ? 8 : 0);

    #pragma unroll
    for (int nb = 0; nb < NOUT / 16; ++nb) {
        const int col = nb * 16 + l16;
        // B fragments: same K order as A (ISA 16-bit table layout), column = lane%16
        H16 b0, b1;
        #pragma unroll
        for (int v = 0; v < 8; ++v) {
            const int k0 = (v < 4 ? 2 * v : 2 * v + 8) + (halfsel ? 8 : 0);
            b0.u[v] = (k0 < KIN)      ? *(const unsigned int*)&sWt[col * 64 + k0]      : 0u;
            b1.u[v] = (k0 + 32 < KIN) ? *(const unsigned int*)&sWt[col * 64 + k0 + 32] : 0u;
        }
        v8f acc = {};
        acc = __builtin_amdgcn_wmma_f32_16x16x32_f16(false, a0.v, false, b0.v,
                                                     (short)0, acc, false, false);
        if (KIN > 32)
            acc = __builtin_amdgcn_wmma_f32_16x16x32_f16(false, a1.v, false, b1.v,
                                                         (short)0, acc, false, false);
        // epilogue: +bias, relu, store, stats (C layout: VGPR r -> rows r / r+8)
        const float bcol = sBias[col];                  // one DS load per nb
        // single 64-bit base address per lane; stores use immediate offsets
        _Float16* __restrict__ baseh = F16OUT ? (outh + (long)row0 * NOUT + col) : (_Float16*)0;
        float*    __restrict__ basef = F16OUT ? (float*)0 : (outf + (long)row0 * NOUT + col);
        float lsum = 0.0f, lsq = 0.0f;
        #pragma unroll
        for (int r = 0; r < 8; ++r) {
            float val = acc[r] + bcol;
            if (RELU) val = fmaxf(val, 0.0f);
            if (!TAIL) {
                if (F16OUT) baseh[r * NOUT] = (_Float16)val;
                else        basef[r * NOUT] = val;
                if (STATS) { lsum += val; lsq += val * val; }
            } else {
                const int row = row0 + r;
                if (row < M) {
                    if (F16OUT) baseh[r * NOUT] = (_Float16)val;
                    else        basef[r * NOUT] = val;
                }
                if (STATS) {
                    float vv = (row < M) ? val : 0.0f;   // branch-free select
                    lsum += vv; lsq += vv * vv;
                }
            }
        }
        if (STATS) {
            atomicAdd(&sSum[col], lsum);
            atomicAdd(&sSumSq[col], lsq);
        }
    }
    if (STATS) {
        __syncthreads();
        if (tid < NOUT) {
            atomicAdd(&statsOut[tid], sSum[tid]);
            atomicAdd(&statsOut[64 + tid], sSumSq[tid]);
        }
    }
}

// helper: launch main (guard-free) + tail instance of one layer
template <int KIN, int NOUT, bool RELU, bool INF32, bool F16OUT, bool STATS>
static void launch_layer(const float* xf, const _Float16* xh, const float* W,
                         const float* bias, const float* g, const float* be,
                         const float* statsIn, float* statsOut,
                         _Float16* outh, float* outf, hipStream_t stream) {
    aff_gemm_kernel<KIN, NOUT, RELU, INF32, F16OUT, STATS, false>
        <<<MAINROWS / MTILE, 256, 0, stream>>>(xf, xh, W, bias, g, be, statsIn,
                                               statsOut, outh, outf, NPAIR, 0);
    aff_gemm_kernel<KIN, NOUT, RELU, INF32, F16OUT, STATS, true>
        <<<1, 256, 0, stream>>>(xf, xh, W, bias, g, be, statsIn,
                                statsOut, outh, outf, NPAIR, MAINROWS);
}

// ---------------------------------------------------------------------------
extern "C" void kernel_launch(void* const* d_in, const int* in_sizes, int n_in,
                              void* d_out, int out_size, void* d_ws, size_t ws_size,
                              hipStream_t stream) {
    (void)in_sizes; (void)n_in; (void)out_size; (void)ws_size;
    // inputs (setup_inputs dict order; aff_params flattened in insertion order)
    const float* cls[3] = { (const float*)d_in[0], (const float*)d_in[4], (const float*)d_in[8] };
    const float* loc[3] = { (const float*)d_in[1], (const float*)d_in[5], (const float*)d_in[9] };
    const float* obj[3] = { (const float*)d_in[3], (const float*)d_in[7], (const float*)d_in[11] };
    const float* det_boxes = (const float*)d_in[12];
    const float* id_embeds = (const float*)d_in[13];
    const float* ref_boxes = (const float*)d_in[14];
    const float* ref_embeds = (const float*)d_in[15];
    const float* w[4]  = { (const float*)d_in[16], (const float*)d_in[18],
                           (const float*)d_in[20], (const float*)d_in[22] };
    const float* bi[4] = { (const float*)d_in[17], (const float*)d_in[19],
                           (const float*)d_in[21], (const float*)d_in[23] };
    const float* g[4]  = { (const float*)d_in[24], (const float*)d_in[26],
                           (const float*)d_in[28], (const float*)d_in[30] };
    const float* be[4] = { (const float*)d_in[25], (const float*)d_in[27],
                           (const float*)d_in[29], (const float*)d_in[31] };

    float* out = (float*)d_out;
    float* scores = out;                                    // [16, 21504, 80]
    float* boxes  = out + (long)16 * NANCHORS * 80;         // [16, 21504, 4]
    float* aff    = boxes + (long)16 * NANCHORS * 4;        // [16, 300, 300, 64]

    // workspace layout (bytes)
    char* ws = (char*)d_ws;
    float*    dn    = (float*)(ws + 0);                     // 4800*64 f32
    float*    rn    = (float*)(ws + 1228800);               // 4800*64 f32
    float*    meta  = (float*)(ws + 2457600);               // 9600*3 f32
    float*    stats = (float*)(ws + 2572800);               // 16*4*128 f32
    float*    feats = (float*)(ws + 2605568);               // 90112*2 f32
    _Float16* h1    = (_Float16*)(ws + 3326464);            // 90112*32 f16
    _Float16* h2    = (_Float16*)(ws + 9093632);            // 90112*32 f16
    _Float16* h3    = (_Float16*)(ws + 14860800);           // 90112*64 f16

    zero_stats_kernel<<<1, 256, 0, stream>>>(stats);

    // decode levels: (H, stride, anchor offset)
    const int   Hs[3]  = { 128, 64, 32 };
    const float Ss[3]  = { 8.0f, 16.0f, 32.0f };
    const int   Ao[3]  = { 0, 16384, 20480 };
    for (int l = 0; l < 3; ++l) {
        dim3 grid(Hs[l] * Hs[l] / 64, 16);
        decode_kernel<<<grid, 256, 0, stream>>>(cls[l], loc[l], obj[l],
                                                scores, boxes, Hs[l], Ss[l], Ao[l]);
    }

    norm_embed_kernel<<<1200, 256, 0, stream>>>(id_embeds, ref_embeds, dn, rn);
    boxmeta_kernel<<<38, 256, 0, stream>>>(det_boxes, ref_boxes, meta);

    for (int b = 0; b < 16; ++b) {
        float* st = stats + (long)b * 512;                  // 4 stages x 128
        feats_kernel<<<352, 256, 0, stream>>>(dn, rn, meta, feats, st + 0 * 128, b);
        // L1: BN0 + Lin(2,32) + ReLU -> h1 (f16), stats for BN1
        launch_layer<2, 32, true, true, true, true>(
            feats, (const _Float16*)nullptr, w[0], bi[0], g[0], be[0],
            st + 0 * 128, st + 1 * 128, h1, (float*)nullptr, stream);
        // L2: BN1 + Lin(32,32) + ReLU -> h2 (f16), stats for BN2
        launch_layer<32, 32, true, false, true, true>(
            (const float*)nullptr, h1, w[1], bi[1], g[1], be[1],
            st + 1 * 128, st + 2 * 128, h2, (float*)nullptr, stream);
        // L3: BN2 + Lin(32,64) + ReLU -> h3 (f16), stats for BN3
        launch_layer<32, 64, true, false, true, true>(
            (const float*)nullptr, h2, w[2], bi[2], g[2], be[2],
            st + 2 * 128, st + 3 * 128, h3, (float*)nullptr, stream);
        // L4: BN3 + Lin(64,64) -> aff[b] (f32, final output)
        launch_layer<64, 64, false, false, false, false>(
            (const float*)nullptr, h3, w[3], bi[3], g[3], be[3],
            st + 3 * 128, (float*)nullptr, (_Float16*)nullptr,
            aff + (long)b * NPAIR * 64, stream);
    }
}
